// Instance_self_disentangle_atten_66700842107137
// MI455X (gfx1250) — matrix operand order repaired and checked by reading, not compiled, not run on hardware
//
#include <hip/hip_runtime.h>
#include <hip/hip_bf16.h>

typedef __attribute__((ext_vector_type(16))) __bf16 v16bf;
typedef __attribute__((ext_vector_type(8)))  __bf16 v8bf;
typedef __attribute__((ext_vector_type(8)))  float  v8f;

#define DIM   256
#define NSEQ  512
#define BATCH 16
#define HEADS 8
#define DHEAD 32

static __device__ __forceinline__ __bf16 f2bf(float f) {
  // round-to-nearest-even f32 -> bf16
  unsigned u = __builtin_bit_cast(unsigned, f);
  unsigned r = (u + 0x7FFFu + ((u >> 16) & 1u)) >> 16;
  unsigned short s = (unsigned short)r;
  return __builtin_bit_cast(__bf16, s);
}

__global__ void cvt_f32_to_bf16(const float* __restrict__ src,
                                __bf16* __restrict__ dst, int n) {
  int i = blockIdx.x * blockDim.x + threadIdx.x;
  if (i < n) dst[i] = f2bf(src[i]);
}

// ---------------------------------------------------------------------------
// Projections: out[r,c] = sum_p src[r,p] * W[c,p] + bias[c]
// pj==0: q (f32, centered later)   pj==1: k (bf16 row-major)
// pj==2: v (bf16 TRANSPOSED: vtb[(b*8+h)*32+d][n])
// One wave per 16-row tile; A (x/y rows) converted to bf16 in-register.
// ---------------------------------------------------------------------------
__global__ __launch_bounds__(32) void proj_kernel(
    const float* __restrict__ x, const float* __restrict__ y,
    const __bf16* __restrict__ Wqb, const __bf16* __restrict__ Wkb,
    const __bf16* __restrict__ Wvb,
    const float* __restrict__ bq, const float* __restrict__ bk,
    const float* __restrict__ bv,
    float* __restrict__ qf, __bf16* __restrict__ kbm, __bf16* __restrict__ vtb) {
  const int rt   = blockIdx.x;        // 0..511 row tiles of 16 tokens
  const int pj   = blockIdx.y;        // 0=q 1=k 2=v
  const int lane = threadIdx.x;
  const int row  = lane & 15;
  const int hi   = lane >> 4;
  const int kA   = hi ? 16 : 0;       // A k-chunk base
  const int kB   = hi ? 16 : 0;       // B k base
  const int r    = rt * 16 + row;

  const float*  src  = (pj == 0) ? x : y;
  const __bf16* W    = (pj == 0) ? Wqb : (pj == 1) ? Wkb : Wvb;
  const float*  bias = (pj == 0) ? bq : (pj == 1) ? bk : bv;

  // preload A: 8 k-steps of the 16x32 bf16 A tile
  v16bf a[8];
#pragma unroll
  for (int ks = 0; ks < 8; ++ks) {
    const float* p0 = src + (size_t)r * DIM + ks * 32 + kA;
#pragma unroll
    for (int e = 0; e < 8; ++e) {
      a[ks][e]     = f2bf(p0[e]);
      a[ks][e + 8] = f2bf(p0[e + 8]);
    }
  }

  for (int ct = 0; ct < 16; ++ct) {
    v8f c = {};
#pragma unroll
    for (int ks = 0; ks < 8; ++ks) {
      const __bf16* bp = W + (size_t)(ct * 16 + row) * DIM + ks * 32 + kB;
      v16bf b = *(const v16bf*)bp;
      c = __builtin_amdgcn_wmma_f32_16x16x32_bf16(false, a[ks], false, b,
                                                  (short)0, c, false, false);
    }
    const int col = ct * 16 + row;
    const float bs = bias[col];
#pragma unroll
    for (int v = 0; v < 8; ++v) {
      const int rq  = v + 8 * hi;
      const int tok = rt * 16 + rq;
      const float val = c[v] + bs;
      if (pj == 0) {
        qf[(size_t)tok * DIM + col] = val;
      } else if (pj == 1) {
        kbm[(size_t)tok * DIM + col] = f2bf(val);
      } else {
        const int bi = tok >> 9, n = tok & 511;
        const int h = col >> 5, d = col & 31;
        vtb[((size_t)((bi * 8 + h) * DHEAD + d)) * NSEQ + n] = f2bf(val);
      }
    }
  }
}

// mean of q over the 512 tokens of each batch, per channel
__global__ void qmean_kernel(const float* __restrict__ qf,
                             float* __restrict__ meanq) {
  const int b = blockIdx.x, c = threadIdx.x;
  float s = 0.f;
  for (int n = 0; n < NSEQ; ++n) s += qf[((size_t)b * NSEQ + n) * DIM + c];
  meanq[b * DIM + c] = s * (1.0f / NSEQ);
}

__global__ void centerq_kernel(const float* __restrict__ qf,
                               const float* __restrict__ meanq,
                               __bf16* __restrict__ qcb) {
  size_t i = (size_t)blockIdx.x * blockDim.x + threadIdx.x;  // 8192*256 total
  int b = (int)(i >> 17);  // / (512*256)
  int c = (int)(i & 255);
  qcb[i] = f2bf(qf[i] - meanq[b * DIM + c]);
}

// m projection + softmax over the 512 tokens: sm[b*8+h][n]
__global__ __launch_bounds__(512) void msoftmax_kernel(
    const float* __restrict__ x, const float* __restrict__ Wm,
    const float* __restrict__ bm, float* __restrict__ sm) {
  const int b = blockIdx.x >> 3, h = blockIdx.x & 7;
  const int n = threadIdx.x;
  const float* xr = x + ((size_t)b * NSEQ + n) * DIM;
  const float* wr = Wm + h * DIM;
  float acc = bm[h];
  for (int p = 0; p < DIM; ++p) acc += xr[p] * wr[p];
  __shared__ float red[512];
  red[n] = acc; __syncthreads();
  for (int s = 256; s > 0; s >>= 1) {
    if (n < s) red[n] = fmaxf(red[n], red[n + s]);
    __syncthreads();
  }
  const float mx = red[0]; __syncthreads();
  const float e = __expf(acc - mx);
  red[n] = e; __syncthreads();
  for (int s = 256; s > 0; s >>= 1) {
    if (n < s) red[n] += red[n + s];
    __syncthreads();
  }
  sm[(size_t)blockIdx.x * NSEQ + n] = e / red[0];
}

// ---------------------------------------------------------------------------
// Attention: one wave per (b,h, 16-query block).
// S = Qc K^T / 16 -> LDS; row softmax (+ sm[key]); ans = P V via WMMA.
// ---------------------------------------------------------------------------
__global__ __launch_bounds__(32) void attn_kernel(
    const __bf16* __restrict__ qcb, const __bf16* __restrict__ kbm,
    const __bf16* __restrict__ vtb, const float* __restrict__ sm,
    float* __restrict__ ansf) {
  __shared__ float sc[16][NSEQ];               // 32 KB
  const int bh = blockIdx.x;                   // b*8+h
  const int b = bh >> 3, h = bh & 7;
  const int qt = blockIdx.y;                   // 0..31
  const int lane = threadIdx.x;
  const int row  = lane & 15;
  const int hi   = lane >> 4;
  const int kA   = hi ? 16 : 0;
  const int kB   = hi ? 16 : 0;

  // A = centered-Q tile [16 queries x dq=32] (single K step)
  v16bf aq;
  {
    const __bf16* p =
        qcb + ((size_t)(b * NSEQ + qt * 16 + row)) * DIM + h * DHEAD + kA;
    v8bf lo  = *(const v8bf*)p;
    v8bf hiv = *(const v8bf*)(p + 8);
    aq = __builtin_shufflevector(lo, hiv, 0, 1, 2, 3, 4, 5, 6, 7, 8, 9, 10, 11,
                                 12, 13, 14, 15);
  }

  const float scale = 0.0625f;  // 1/sqrt(DIM_Q)
  for (int kt = 0; kt < 32; ++kt) {
    const int key = kt * 16 + row;             // B column for this lane
    const __bf16* bp =
        kbm + ((size_t)(b * NSEQ + key)) * DIM + h * DHEAD + kB;
    v16bf bk = *(const v16bf*)bp;
    v8f c = {};
    c = __builtin_amdgcn_wmma_f32_16x16x32_bf16(false, aq, false, bk, (short)0,
                                                c, false, false);
#pragma unroll
    for (int v = 0; v < 8; ++v) sc[v + 8 * hi][key] = c[v] * scale;
  }

  // row softmax over 512 keys; lane pair (L, L^16) splits the row
  const int c0 = hi * 256;
  float mx = -3.0e38f;
  for (int j = 0; j < 256; ++j) mx = fmaxf(mx, sc[row][c0 + j]);
  mx = fmaxf(mx, __shfl_xor(mx, 16, 32));
  float sum = 0.f;
  for (int j = 0; j < 256; ++j) sum += __expf(sc[row][c0 + j] - mx);
  sum += __shfl_xor(sum, 16, 32);
  const float inv = 1.0f / sum;
  const float* smr = sm + (size_t)bh * NSEQ;
  for (int j = 0; j < 256; ++j) {
    const int key = c0 + j;
    sc[row][key] = __expf(sc[row][key] - mx) * inv + smr[key];
  }

  // PV: ans[16 x 32] = P[16 x 512] @ V[512 x 32]
#pragma unroll
  for (int dt = 0; dt < 2; ++dt) {
    const int d = dt * 16 + row;               // B column (dv index)
    v8f c = {};
    for (int kt = 0; kt < 32; ++kt) {
      v16bf pa;
#pragma unroll
      for (int e = 0; e < 8; ++e) {
        pa[e]     = f2bf(sc[row][kt * 32 + kA + e]);
        pa[e + 8] = f2bf(sc[row][kt * 32 + kA + 8 + e]);
      }
      const __bf16* vp =
          vtb + ((size_t)(bh * DHEAD + d)) * NSEQ + kt * 32 + kB;
      v16bf vb = *(const v16bf*)vp;
      c = __builtin_amdgcn_wmma_f32_16x16x32_bf16(false, pa, false, vb,
                                                  (short)0, c, false, false);
    }
#pragma unroll
    for (int v = 0; v < 8; ++v) {
      const int n = qt * 16 + v + 8 * hi;
      ansf[((size_t)bh * NSEQ + n) * DHEAD + d] = c[v];
    }
  }
}

// raw-reinterpret view/permute: out slot (b',n,h'*32+d) <- flat head h'*16+b'
__global__ void remap_kernel(const float* __restrict__ ansf,
                             __bf16* __restrict__ ansb) {
  size_t i = (size_t)blockIdx.x * blockDim.x + threadIdx.x;  // 8192*256 total
  const int bp = (int)(i >> 17);
  const int n  = (int)((i >> 8) & 511);
  const int c  = (int)(i & 255);
  const int hp = c >> 5, d = c & 31;
  const int src = hp * 16 + bp;  // == bb*8+hh
  ansb[i] = f2bf(ansf[((size_t)src * NSEQ + n) * DHEAD + d]);
}

// fc + residual + LayerNorm, one wave per 16-row tile
__global__ __launch_bounds__(32) void fcln_kernel(
    const __bf16* __restrict__ ansb, const __bf16* __restrict__ Wfcb,
    const float* __restrict__ bfc, const float* __restrict__ x,
    const float* __restrict__ gamma, const float* __restrict__ beta,
    float* __restrict__ out) {
  __shared__ float hb[16][DIM];                // 16 KB
  const int rt = blockIdx.x;
  const int lane = threadIdx.x;
  const int row = lane & 15;
  const int hi = lane >> 4;
  const int kA = hi ? 16 : 0;
  const int kB = hi ? 16 : 0;
  const int r = rt * 16 + row;

  v16bf a[8];
#pragma unroll
  for (int ks = 0; ks < 8; ++ks) {
    const __bf16* p = ansb + (size_t)r * DIM + ks * 32 + kA;
    v8bf lo  = *(const v8bf*)p;
    v8bf hiv = *(const v8bf*)(p + 8);
    a[ks] = __builtin_shufflevector(lo, hiv, 0, 1, 2, 3, 4, 5, 6, 7, 8, 9, 10,
                                    11, 12, 13, 14, 15);
  }
  for (int ct = 0; ct < 16; ++ct) {
    v8f c = {};
#pragma unroll
    for (int ks = 0; ks < 8; ++ks) {
      const __bf16* bp = Wfcb + (size_t)(ct * 16 + row) * DIM + ks * 32 + kB;
      v16bf b = *(const v16bf*)bp;
      c = __builtin_amdgcn_wmma_f32_16x16x32_bf16(false, a[ks], false, b,
                                                  (short)0, c, false, false);
    }
    const int col = ct * 16 + row;
    const float bb = bfc[col];
#pragma unroll
    for (int v = 0; v < 8; ++v) {
      const int rq = v + 8 * hi;
      hb[rq][col] = c[v] + bb + x[((size_t)(rt * 16 + rq)) * DIM + col];
    }
  }
  // LayerNorm: lane pair (L, L^16) splits the 256 channels of its row
  const int c0 = hi * 128;
  float s = 0.f, ss = 0.f;
  for (int j = 0; j < 128; ++j) {
    const float v = hb[row][c0 + j];
    s += v; ss += v * v;
  }
  s  += __shfl_xor(s, 16, 32);
  ss += __shfl_xor(ss, 16, 32);
  const float mu = s * (1.0f / DIM);
  const float var = ss * (1.0f / DIM) - mu * mu;
  const float rstd = rsqrtf(var + 1e-5f);
  for (int j = 0; j < 128; ++j) {
    const int cc = c0 + j;
    const float v = (hb[row][cc] - mu) * rstd;
    out[((size_t)(rt * 16 + row)) * DIM + cc] = v * gamma[cc] + beta[cc];
  }
}

extern "C" void kernel_launch(void* const* d_in, const int* in_sizes, int n_in,
                              void* d_out, int out_size, void* d_ws,
                              size_t ws_size, hipStream_t stream) {
  (void)in_sizes; (void)n_in; (void)out_size; (void)ws_size;
  const float* x    = (const float*)d_in[0];
  const float* y    = (const float*)d_in[1];
  const float* Wq   = (const float*)d_in[2];
  const float* bq   = (const float*)d_in[3];
  const float* Wk   = (const float*)d_in[4];
  const float* bk   = (const float*)d_in[5];
  const float* Wv   = (const float*)d_in[6];
  const float* bv   = (const float*)d_in[7];
  const float* Wm   = (const float*)d_in[8];
  const float* bm   = (const float*)d_in[9];
  const float* Wfc  = (const float*)d_in[10];
  const float* bfc  = (const float*)d_in[11];
  const float* gam  = (const float*)d_in[12];
  const float* bet  = (const float*)d_in[13];
  float* out = (float*)d_out;

  char* ws = (char*)d_ws;
  // region 0 (8 MB): qf (proj->centerq), then reused as ansf (attn->remap)
  float*  qf    = (float*)(ws + 0);
  float*  ansf  = (float*)(ws + 0);
  __bf16* qcb   = (__bf16*)(ws + 8388608);
  __bf16* kbm   = (__bf16*)(ws + 12582912);
  __bf16* vtb   = (__bf16*)(ws + 16777216);
  __bf16* ansb  = (__bf16*)(ws + 20971520);
  float*  sm    = (float*)(ws + 25165824);
  float*  meanq = (float*)(ws + 25427968);
  __bf16* Wqb   = (__bf16*)(ws + 25444352);
  __bf16* Wkb   = (__bf16*)(ws + 25575424);
  __bf16* Wvb   = (__bf16*)(ws + 25706496);
  __bf16* Wfcb  = (__bf16*)(ws + 25837568);

  cvt_f32_to_bf16<<<256, 256, 0, stream>>>(Wq, Wqb, 65536);
  cvt_f32_to_bf16<<<256, 256, 0, stream>>>(Wk, Wkb, 65536);
  cvt_f32_to_bf16<<<256, 256, 0, stream>>>(Wv, Wvb, 65536);
  cvt_f32_to_bf16<<<256, 256, 0, stream>>>(Wfc, Wfcb, 65536);
  proj_kernel<<<dim3(512, 3), 32, 0, stream>>>(x, y, Wqb, Wkb, Wvb, bq, bk, bv,
                                               qf, kbm, vtb);
  qmean_kernel<<<16, 256, 0, stream>>>(qf, meanq);
  centerq_kernel<<<8192, 256, 0, stream>>>(qf, meanq, qcb);
  msoftmax_kernel<<<128, 512, 0, stream>>>(x, Wm, bm, sm);
  attn_kernel<<<dim3(128, 32), 32, 0, stream>>>(qcb, kbm, vtb, sm, ansf);
  remap_kernel<<<8192, 256, 0, stream>>>(ansf, ansb);
  fcln_kernel<<<512, 32, 0, stream>>>(ansb, Wfcb, bfc, x, gam, bet, out);
}
